// GraphUNet_7748121002348
// MI455X (gfx1250) — matrix operand order repaired and checked by reading, not compiled, not run on hardware
//
#include <hip/hip_runtime.h>
#include <hip/hip_bf16.h>
#include <cstdint>

// ---------------- problem constants (match reference) ----------------
#define NNODES 4096
#define NEDGES 65536
#define CH     128
#define DEPTHN 3

// ---------------- WMMA types / tiling ----------------
typedef __attribute__((ext_vector_type(16))) __bf16 v16bf;
typedef __attribute__((ext_vector_type(8)))  __bf16 v8bf;
typedef __attribute__((ext_vector_type(8)))  float  v8f;

#define BM 128
#define BN 128
#define BK 32
#define A_STRIDE 40   // padded LDS row stride (bf16 elems), keeps 16B alignment
#define B_STRIDE 40

// =====================================================================
// Build dense adjacency: A[src*N + tgt] += 1 per edge (duplicates sum).
// =====================================================================
__global__ void k_build_adj(const int* __restrict__ ei, float* __restrict__ A) {
  int e = blockIdx.x * blockDim.x + threadIdx.x;
  if (e < NEDGES) {
    int s = ei[e];
    int t = ei[NEDGES + e];
    atomicAdd(&A[(size_t)s * NNODES + t], 1.0f);
  }
}

// =====================================================================
// deg[t] = colsum(A')[t] with improved self-loops (+2 where diag==0);
// store dinv = rsqrt(deg).
// =====================================================================
__global__ void k_colsum_deg(const float* __restrict__ A, int n,
                             float* __restrict__ dinv) {
  int t = blockIdx.x * blockDim.x + threadIdx.x;
  if (t >= n) return;
  float d = 0.0f;
  for (int s = 0; s < n; ++s) d += A[(size_t)s * n + t];
  if (A[(size_t)t * n + t] == 0.0f) d += 2.0f;
  dinv[t] = rsqrtf(d);
}

// =====================================================================
// out_bf16[t][s] = bf16( A'[s][t] * dinv[s] * dinv[t] )   (An.T, normalized)
// A'[s][t] = A[s][t], except diag: if A[t][t]==0 -> 2.0 (improved GCN).
// LDS-tiled transpose so both global accesses are coalesced.
// =====================================================================
__global__ void k_norm_transpose(const float* __restrict__ A,
                                 const float* __restrict__ dinv, int n,
                                 __bf16* __restrict__ out) {
  __shared__ float tile[32][33];
  int bx = blockIdx.x * 32, by = blockIdx.y * 32;
  for (int i = threadIdx.y; i < 32; i += 8) {
    int s = by + i, t = bx + threadIdx.x;
    float v = A[(size_t)s * n + t];
    if (s == t && v == 0.0f) v = 2.0f;
    tile[i][threadIdx.x] = v * dinv[s] * dinv[t];
  }
  __syncthreads();
  for (int i = threadIdx.y; i < 32; i += 8) {
    int t = bx + i, s = by + threadIdx.x;
    out[(size_t)t * n + s] = (__bf16)tile[threadIdx.x][i];
  }
}

// =====================================================================
// f32 -> bf16 conversion; optionally force diagonal to 1.0 (_augment_adj
// does A*(1-eye)+eye before the dense square).
// =====================================================================
__global__ void k_convert_bf16(__bf16* __restrict__ dst,
                               const float* __restrict__ src,
                               int rows, int cols, int diag_one) {
  int idx = blockIdx.x * blockDim.x + threadIdx.x;
  if (idx >= rows * cols) return;
  int r = idx / cols, c = idx - r * cols;
  float v = src[idx];
  if (diag_one && r == c) v = 1.0f;
  dst[idx] = (__bf16)v;
}

// =====================================================================
// bf16 WMMA GEMM:  C(f32, MxN) = A(bf16, MxK row-major) * B(bf16, KxN rm)
// Epilogue: + bias[n] (optional), ReLU (optional), zero global diagonal
// (optional, for _augment_adj).
//
// Block tile 128x128, 256 threads = 8 wave32 waves arranged 4(M) x 2(N);
// each wave owns a 32x64 region = 2x4 fragments -> 8 x
// v_wmma_f32_16x16x32_bf16 per 32-wide K-step off 12 b128 LDS loads.
// A tile is staged with CDNA5 GLOBAL_LOAD_ASYNC_TO_LDS_B128 (ASYNCcnt);
// B tile is loaded in coalesced 16B row chunks and transposed into padded
// n-major LDS via register scatter, so B-fragment reads are contiguous
// b128 loads. Requires M%128==0, N%128==0, K%32==0 (true for all call
// sites: dims are 512/1024/2048/4096 and C=128).
// =====================================================================
__global__ __launch_bounds__(256) void k_gemm_bf16(
    float* __restrict__ Cc, const __bf16* __restrict__ A,
    const __bf16* __restrict__ B, int M, int Nn, int K,
    const float* __restrict__ bias, int relu, int zero_diag) {
  __shared__ __bf16 As[BM * A_STRIDE];
  __shared__ __bf16 Bs[BN * B_STRIDE];   // transposed: Bs[n][k]

  const int tid  = threadIdx.x;
  const int lane = tid & 31;
  const int wv   = tid >> 5;
  const int wm   = wv >> 1;      // 0..3 (M dir)
  const int wn   = wv & 1;       // 0..1 (N dir)
  const int hf   = lane >> 4;    // lane half (0/1)
  const int l15  = lane & 15;

  const int bM = blockIdx.y * BM;
  const int bN = blockIdx.x * BN;

  v8f acc[2][4] = {};

  for (int k0 = 0; k0 < K; k0 += BK) {
    // ---- stage A tile (128x32) via async global->LDS DMA: 2 x 16B/thread
#pragma unroll
    for (int c = 0; c < 2; ++c) {
      int ch  = tid * 2 + c;        // 0..511 16B chunks
      int r   = ch >> 2;            // 0..127
      int col = (ch & 3) * 8;       // 0,8,16,24
      uint32_t lds = (uint32_t)(uintptr_t)&As[r * A_STRIDE + col];
      const __bf16* g = A + (size_t)(bM + r) * K + (k0 + col);
      asm volatile("global_load_async_to_lds_b128 %0, %1, off"
                   :: "v"(lds), "v"((uint64_t)(uintptr_t)g)
                   : "memory");
    }
    // ---- stage B tile (32x128) transposed into Bs[n][k].
    // Two independent coalesced 16B row-chunk loads per thread, then a
    // register->LDS 2B scatter (no load->store serialization chain).
    {
      v8bf vb[2];
#pragma unroll
      for (int it = 0; it < 2; ++it) {
        int ch = it * 256 + tid;      // 0..511 8-elem chunks
        int kk = ch >> 4;             // 0..31
        int nn = (ch & 15) * 8;       // 0,8,...,120
        vb[it] = *(const v8bf*)&B[(size_t)(k0 + kk) * Nn + (bN + nn)];
      }
#pragma unroll
      for (int it = 0; it < 2; ++it) {
        int ch = it * 256 + tid;
        int kk = ch >> 4;
        int nn = (ch & 15) * 8;
#pragma unroll
        for (int j = 0; j < 8; ++j)
          Bs[(nn + j) * B_STRIDE + kk] = vb[it][j];
      }
    }
    // prefetch next B tile into cache (speculative, LOADcnt-free)
    if (k0 + BK < K)
      __builtin_prefetch(&B[(size_t)(k0 + BK + (tid >> 7)) * Nn + bN + (tid & 127)], 0, 1);
    asm volatile("s_wait_asynccnt 0x0" ::: "memory");
    __syncthreads();

    // A fragment: lane l15 = row m; element e -> K=(e<8?e:e+8)+hf*8
    v16bf af[2];
#pragma unroll
    for (int tr = 0; tr < 2; ++tr) {
      const __bf16* p = &As[(wm * 32 + tr * 16 + l15) * A_STRIDE + hf * 8];
      v8bf lo = *(const v8bf*)p;
      v8bf hi = *(const v8bf*)(p + 16);
      v16bf f;
#pragma unroll
      for (int q = 0; q < 8; ++q) { f[q] = lo[q]; f[q + 8] = hi[q]; }
      af[tr] = f;
    }
    // B fragment: lane l15 = col n; element e -> K=(e<8?e:e+8)+hf*8
    v16bf bfm[4];
#pragma unroll
    for (int tc = 0; tc < 4; ++tc) {
      const __bf16* p = &Bs[(wn * 64 + tc * 16 + l15) * B_STRIDE + hf * 8];
      v8bf lo = *(const v8bf*)p;
      v8bf hi = *(const v8bf*)(p + 16);
      v16bf f;
#pragma unroll
      for (int q = 0; q < 8; ++q) { f[q] = lo[q]; f[q + 8] = hi[q]; }
      bfm[tc] = f;
    }
#pragma unroll
    for (int tr = 0; tr < 2; ++tr)
#pragma unroll
      for (int tc = 0; tc < 4; ++tc)
        acc[tr][tc] = __builtin_amdgcn_wmma_f32_16x16x32_bf16(
            false, af[tr], false, bfm[tc], (short)0, acc[tr][tc], false, false);
    __syncthreads();
  }

  // D layout: lanes 0-15 -> N=lane, M=r; lanes 16-31 -> N=lane-16, M=8+r
#pragma unroll
  for (int tr = 0; tr < 2; ++tr) {
#pragma unroll
    for (int tc = 0; tc < 4; ++tc) {
      int gn = bN + wn * 64 + tc * 16 + l15;
      float bv = bias ? bias[gn] : 0.0f;
#pragma unroll
      for (int r = 0; r < 8; ++r) {
        int gm = bM + wm * 32 + tr * 16 + hf * 8 + r;
        float v = acc[tr][tc][r] + bv;
        if (relu) v = fmaxf(v, 0.0f);
        if (zero_diag && gm == gn) v = 0.0f;
        Cc[(size_t)gm * Nn + gn] = v;
      }
    }
  }
}

// =====================================================================
// TopK pooling pieces
// =====================================================================
__global__ void k_scores(const float* __restrict__ x,
                         const float* __restrict__ w, int n,
                         float* __restrict__ score) {
  int i = blockIdx.x * blockDim.x + threadIdx.x;
  if (i >= n) return;
  float dot = 0.0f, wn = 0.0f;
  for (int c = 0; c < CH; ++c) {
    dot += x[(size_t)i * CH + c] * w[c];
    wn  += w[c] * w[c];
  }
  score[i] = tanhf(dot * rsqrtf(wn));
}

// Exact, stable top-k: rank(i) = #{j : s[j]>s[i] or (s[j]==s[i] && j<i)}.
// Ranks form a bijection, so perm[rank]=i fills ranks 0..k-1 exactly once,
// sorted descending with lax.top_k's lower-index-first tie break.
__global__ void k_rank_select(const float* __restrict__ score, int n, int k,
                              int* __restrict__ perm) {
  int i = blockIdx.x * blockDim.x + threadIdx.x;
  if (i >= n) return;
  float si = score[i];
  int rank = 0;
  for (int j = 0; j < n; ++j) {
    float sj = score[j];
    rank += (sj > si) || (sj == si && j < i);
  }
  if (rank < k) perm[rank] = i;
}

__global__ void k_gather_x(float* __restrict__ xp, const float* __restrict__ x,
                           const int* __restrict__ perm,
                           const float* __restrict__ score, int k) {
  int idx = blockIdx.x * blockDim.x + threadIdx.x;
  if (idx >= k * CH) return;
  int r = idx / CH, c = idx - r * CH;
  int p = perm[r];
  xp[idx] = x[(size_t)p * CH + c] * score[p];
}

__global__ void k_gather_A(float* __restrict__ Ap, const float* __restrict__ Caug,
                           const int* __restrict__ perm, int n, int k) {
  int idx = blockIdx.x * blockDim.x + threadIdx.x;
  if (idx >= k * k) return;
  int r = idx / k, c = idx - r * k;
  Ap[idx] = Caug[(size_t)perm[r] * n + perm[c]];
}

// dst[perm[r]][:] += xs[r][:]  (unpool scatter; perm rows are unique)
__global__ void k_scatter_add(float* __restrict__ dst,
                              const float* __restrict__ xs,
                              const int* __restrict__ perm, int k) {
  int idx = blockIdx.x * blockDim.x + threadIdx.x;
  if (idx >= k * CH) return;
  int r = idx / CH, c = idx - r * CH;
  size_t o = (size_t)perm[r] * CH + c;
  dst[o] += xs[idx];
}

// =====================================================================
// Host-side orchestration
// =====================================================================
namespace {

struct WS {
  float *A0, *CT, *A1, *A2, *A3;
  __bf16 *ABF, *HBF, *XBF, *WBF;
  float *HTMP, *X0, *X1, *X2, *XP, *XB, *XU2, *XU1, *TMP;
  float *DINV, *SCORE;
  int *P1, *P2, *P3;
};

inline void run_gcn(const float* A, int n, const float* x, const float* W,
                    const float* b, float* out, int relu, const WS& w,
                    hipStream_t s) {
  k_colsum_deg<<<n / 256, 256, 0, s>>>(A, n, w.DINV);
  dim3 tg(32, 8), gg(n / 32, n / 32);
  k_norm_transpose<<<gg, tg, 0, s>>>(A, w.DINV, n, w.ABF);
  int nc = n * CH;
  k_convert_bf16<<<(nc + 255) / 256, 256, 0, s>>>(w.XBF, x, n, CH, 0);
  k_convert_bf16<<<(CH * CH + 255) / 256, 256, 0, s>>>(w.WBF, W, CH, CH, 0);
  dim3 g1(CH / BN, n / BM);
  k_gemm_bf16<<<g1, 256, 0, s>>>(w.HTMP, w.XBF, w.WBF, n, CH, CH,
                                 nullptr, 0, 0);
  k_convert_bf16<<<(nc + 255) / 256, 256, 0, s>>>(w.HBF, w.HTMP, n, CH, 0);
  k_gemm_bf16<<<g1, 256, 0, s>>>(out, w.ABF, w.HBF, n, CH, n, b, relu, 0);
}

inline void run_augment(const float* A, int n, const WS& w, hipStream_t s) {
  // B = A with diag forced to 1; CT = B@B with zeroed diagonal
  k_convert_bf16<<<(n * n + 255) / 256, 256, 0, s>>>(w.ABF, A, n, n, 1);
  dim3 g(n / BN, n / BM);
  k_gemm_bf16<<<g, 256, 0, s>>>(w.CT, w.ABF, w.ABF, n, n, n, nullptr, 0, 1);
}

inline void run_pool(const float* x, const float* pw, int n, int k, int* perm,
                     float* xp, float* Anext, const WS& w, hipStream_t s) {
  k_scores<<<n / 256, 256, 0, s>>>(x, pw, n, w.SCORE);
  k_rank_select<<<n / 256, 256, 0, s>>>(w.SCORE, n, k, perm);
  k_gather_x<<<(k * CH + 255) / 256, 256, 0, s>>>(xp, x, perm, w.SCORE, k);
  k_gather_A<<<(k * k + 255) / 256, 256, 0, s>>>(Anext, w.CT, perm, n, k);
}

}  // namespace

extern "C" void kernel_launch(void* const* d_in, const int* in_sizes, int n_in,
                              void* d_out, int out_size, void* d_ws,
                              size_t ws_size, hipStream_t stream) {
  const float* x_in    = (const float*)d_in[0];
  const int*   eidx    = (const int*)d_in[1];
  const float* W_in    = (const float*)d_in[2];
  const float* b_in    = (const float*)d_in[3];
  const float* W_down  = (const float*)d_in[4];
  const float* b_down  = (const float*)d_in[5];
  const float* pool_w  = (const float*)d_in[6];
  const float* W_up    = (const float*)d_in[7];
  const float* b_up    = (const float*)d_in[8];
  float* out = (float*)d_out;

  // -------- workspace layout (bytes) --------
  char* base = (char*)d_ws;
  size_t off = 0;
  auto carve = [&](size_t bytes) {
    void* p = base + off;
    off += (bytes + 255) & ~(size_t)255;
    return p;
  };
  WS w;
  w.A0   = (float*)carve((size_t)NNODES * NNODES * 4);   // 64 MB, persists
  w.CT   = (float*)carve((size_t)NNODES * NNODES * 4);   // 64 MB, augment tmp
  w.A1   = (float*)carve((size_t)2048 * 2048 * 4);
  w.A2   = (float*)carve((size_t)1024 * 1024 * 4);
  w.A3   = (float*)carve((size_t)512 * 512 * 4);
  w.ABF  = (__bf16*)carve((size_t)NNODES * NNODES * 2);  // 32 MB bf16 operand
  w.HBF  = (__bf16*)carve((size_t)NNODES * CH * 2);
  w.XBF  = (__bf16*)carve((size_t)NNODES * CH * 2);
  w.WBF  = (__bf16*)carve((size_t)CH * CH * 2);
  w.HTMP = (float*)carve((size_t)NNODES * CH * 4);
  w.X0   = (float*)carve((size_t)NNODES * CH * 4);
  w.X1   = (float*)carve((size_t)2048 * CH * 4);
  w.X2   = (float*)carve((size_t)1024 * CH * 4);
  w.XP   = (float*)carve((size_t)2048 * CH * 4);
  w.XB   = (float*)carve((size_t)512 * CH * 4);
  w.XU2  = (float*)carve((size_t)1024 * CH * 4);
  w.XU1  = (float*)carve((size_t)2048 * CH * 4);
  w.TMP  = (float*)carve((size_t)NNODES * CH * 4);
  w.DINV = (float*)carve((size_t)NNODES * 4);
  w.SCORE= (float*)carve((size_t)NNODES * 4);
  w.P1   = (int*)carve((size_t)2048 * 4);
  w.P2   = (int*)carve((size_t)1024 * 4);
  w.P3   = (int*)carve((size_t)512 * 4);
  (void)ws_size; (void)in_sizes; (void)n_in; (void)out_size;

  const size_t CC = (size_t)CH * CH;

  // -------- build dense adjacency --------
  hipMemsetAsync(w.A0, 0, (size_t)NNODES * NNODES * 4, stream);
  k_build_adj<<<NEDGES / 256, 256, 0, stream>>>(eidx, w.A0);

  // -------- down path --------
  run_gcn(w.A0, NNODES, x_in, W_in, b_in, w.X0, 1, w, stream);

  // level 0: 4096 -> 2048
  run_augment(w.A0, NNODES, w, stream);
  run_pool(w.X0, pool_w + 0 * CH, NNODES, 2048, w.P1, w.XP, w.A1, w, stream);
  run_gcn(w.A1, 2048, w.XP, W_down + 0 * CC, b_down + 0 * CH, w.X1, 1, w, stream);

  // level 1: 2048 -> 1024
  run_augment(w.A1, 2048, w, stream);
  run_pool(w.X1, pool_w + 1 * CH, 2048, 1024, w.P2, w.XP, w.A2, w, stream);
  run_gcn(w.A2, 1024, w.XP, W_down + 1 * CC, b_down + 1 * CH, w.X2, 1, w, stream);

  // level 2: 1024 -> 512
  run_augment(w.A2, 1024, w, stream);
  run_pool(w.X2, pool_w + 2 * CH, 1024, 512, w.P3, w.XP, w.A3, w, stream);
  run_gcn(w.A3, 512, w.XP, W_down + 2 * CC, b_down + 2 * CH, w.XB, 1, w, stream);

  // -------- up path --------
  // i=0 (j=2): res=X2 (1024), unpool XB via P3, gcn on A2, relu
  hipMemcpyAsync(w.TMP, w.X2, (size_t)1024 * CH * 4,
                 hipMemcpyDeviceToDevice, stream);
  k_scatter_add<<<(512 * CH + 255) / 256, 256, 0, stream>>>(w.TMP, w.XB, w.P3, 512);
  run_gcn(w.A2, 1024, w.TMP, W_up + 0 * CC, b_up + 0 * CH, w.XU2, 1, w, stream);

  // i=1 (j=1): res=X1 (2048), unpool XU2 via P2, gcn on A1, relu
  hipMemcpyAsync(w.TMP, w.X1, (size_t)2048 * CH * 4,
                 hipMemcpyDeviceToDevice, stream);
  k_scatter_add<<<(1024 * CH + 255) / 256, 256, 0, stream>>>(w.TMP, w.XU2, w.P2, 1024);
  run_gcn(w.A1, 2048, w.TMP, W_up + 1 * CC, b_up + 1 * CH, w.XU1, 1, w, stream);

  // i=2 (j=0): res=X0 (4096), unpool XU1 via P1, gcn on A0, NO relu -> out
  hipMemcpyAsync(w.TMP, w.X0, (size_t)NNODES * CH * 4,
                 hipMemcpyDeviceToDevice, stream);
  k_scatter_add<<<(2048 * CH + 255) / 256, 256, 0, stream>>>(w.TMP, w.XU1, w.P1, 2048);
  run_gcn(w.A0, NNODES, w.TMP, W_up + 2 * CC, b_up + 2 * CH, out, 0, w, stream);
}
